// GatedGCNModel_39711267619013
// MI455X (gfx1250) — compile-verified, matrix operand order checked by reading, not run on hardware
//
#include <hip/hip_runtime.h>
#include <hip/hip_bf16.h>
#include <cstdint>
#include <cstddef>

// ---------------------------------------------------------------------------
// GatedGCN forward for MI455X (gfx1250, wave32).
// All GEMM-like stages lower to v_wmma_f32_16x16x32_f16 (f16 in, f32 accum).
// Scratch (d_ws) requirement ~240 MB.
// ---------------------------------------------------------------------------

typedef __attribute__((ext_vector_type(16))) _Float16 v16h;
typedef __attribute__((ext_vector_type(8)))  float    v8f;

#define WMMA(a, b, c) \
  __builtin_amdgcn_wmma_f32_16x16x32_f16(false, (a), false, (b), (short)0, (c), false, false)

// K index pattern for 16-bit A/B fragments (ISA 7.12.2, 16-bit A 16x32):
// lane L: row M = L%16, half-wave hi = L/16 selects K+8 / K+24 groups.
__device__ __forceinline__ int kpat(int v, int hb, int hi) {
  return ((v & 4) ? 16 : 0) + (hi ? 8 : 0) + ((v & 3) << 1) + hb;
}

__device__ __forceinline__ v16h afrag_row16(const _Float16* rowp, int hi, int kbase) {
  v16h a;
#pragma unroll
  for (int v = 0; v < 8; ++v)
#pragma unroll
    for (int hb = 0; hb < 2; ++hb)
      a[2 * v + hb] = rowp[kbase + kpat(v, hb, hi)];
  return a;
}
__device__ __forceinline__ v16h afrag_row32(const float* rowp, int hi, int kbase) {
  v16h a;
#pragma unroll
  for (int v = 0; v < 8; ++v)
#pragma unroll
    for (int hb = 0; hb < 2; ++hb)
      a[2 * v + hb] = (_Float16)rowp[kbase + kpat(v, hb, hi)];
  return a;
}
__device__ __forceinline__ v16h afrag_lds(const _Float16* A, int stride, int m, int hi, int kbase) {
  return afrag_row16(A + (size_t)m * stride, hi, kbase);
}
// B fragment from LDS f16 row-major [K][ncols]; lane L supplies column nbase+L%16.
__device__ __forceinline__ v16h bfrag_lds(const _Float16* B, int ncols, int nbase, int lane, int kbase) {
  int n = nbase + (lane & 15);
  int hi = lane >> 4;
  v16h b;
#pragma unroll
  for (int v = 0; v < 8; ++v)
#pragma unroll
    for (int hb = 0; hb < 2; ++hb)
      b[2 * v + hb] = B[(size_t)(kbase + kpat(v, hb, hi)) * ncols + n];
  return b;
}

__device__ __forceinline__ float sigf(float x) { return 1.f / (1.f + expf(-x)); }
// order-preserving float<->uint mapping for atomic max on floats
__device__ __forceinline__ unsigned ford(float f) {
  unsigned u = __float_as_uint(f);
  return (u & 0x80000000u) ? ~u : (u | 0x80000000u);
}
__device__ __forceinline__ float fordinv(unsigned u) {
  return (u & 0x80000000u) ? __uint_as_float(u & 0x7fffffffu) : __uint_as_float(~u);
}

// ---------------------------------------------------------------------------
// elementwise / reduction kernels
// ---------------------------------------------------------------------------
__global__ void k_zero(float* p, int n) {
  int i = blockIdx.x * blockDim.x + threadIdx.x;
  if (i < n) p[i] = 0.f;
}

__global__ void k_ndir_accum(const float* __restrict__ pos, const int* __restrict__ src,
                             const int* __restrict__ dst, float* __restrict__ ndir, int E_) {
  int e = blockIdx.x * blockDim.x + threadIdx.x;
  if (e >= E_) return;
  int s = src[e], t = dst[e];
  float vx = pos[t * 3 + 0] - pos[s * 3 + 0];
  float vy = pos[t * 3 + 1] - pos[s * 3 + 1];
  float vz = pos[t * 3 + 2] - pos[s * 3 + 2];
  float d = sqrtf(vx * vx + vy * vy + vz * vz + 1e-12f);
  float inv = 1.f / d;
  atomicAdd(&ndir[s * 3 + 0], vx * inv);
  atomicAdd(&ndir[s * 3 + 1], vy * inv);
  atomicAdd(&ndir[s * 3 + 2], vz * inv);
}

__global__ void k_ndir_norm(float* ndir, int NN) {
  int i = blockIdx.x * blockDim.x + threadIdx.x;
  if (i >= NN) return;
  float x = ndir[i * 3], y = ndir[i * 3 + 1], z = ndir[i * 3 + 2];
  float inv = 1.f / (sqrtf(x * x + y * y + z * z) + 1e-8f);
  ndir[i * 3] = x * inv; ndir[i * 3 + 1] = y * inv; ndir[i * 3 + 2] = z * inv;
}

__global__ void k_init_h(const float* __restrict__ x, float* __restrict__ h,
                         _Float16* __restrict__ h16, int NN) {
  int i = blockIdx.x * blockDim.x + threadIdx.x;
  if (i >= NN * 64) return;
  int c = i & 63;
  float v = (c < 4) ? x[(size_t)(i >> 6) * 4 + c] : 0.f;
  h[i] = v; h16[i] = (_Float16)v;
}

__global__ void k_smax_init(unsigned* mord, float* ssum, int n) {
  int i = blockIdx.x * blockDim.x + threadIdx.x;
  if (i < n) { mord[i] = ford(-3.0e38f); ssum[i] = 0.f; }
}
__global__ void k_smax1(const float* logit, const int* dst, unsigned* mord, int E_) {
  int e = blockIdx.x * blockDim.x + threadIdx.x;
  if (e < E_) atomicMax(&mord[dst[e]], ford(logit[e]));
}
__global__ void k_smax2(const float* logit, const int* dst, const unsigned* mord,
                        float* ssum, float* aE, int E_) {
  int e = blockIdx.x * blockDim.x + threadIdx.x;
  if (e >= E_) return;
  float w = expf(logit[e] - fordinv(mord[dst[e]]));
  aE[e] = w;
  atomicAdd(&ssum[dst[e]], w);
}
__global__ void k_smax3(const int* dst, const float* ssum, float* aE, int E_) {
  int e = blockIdx.x * blockDim.x + threadIdx.x;
  if (e < E_) aE[e] = aE[e] / (ssum[dst[e]] + 1e-16f);
}

__global__ void k_message(const float* __restrict__ hW, const _Float16* __restrict__ gate,
                          const int* __restrict__ src, const int* __restrict__ dst,
                          float* __restrict__ agg, int E_) {
  int idx = blockIdx.x * blockDim.x + threadIdx.x;
  if (idx >= E_ * 64) return;
  int e = idx >> 6, c = idx & 63;
  float m = hW[(size_t)src[e] * 64 + c] * (float)gate[idx];
  atomicAdd(&agg[(size_t)dst[e] * 64 + c], m);
}

__global__ void k_bnrelu(float* __restrict__ h, _Float16* __restrict__ h16,
                         const float* g, const float* b, const float* m, const float* v, int NN) {
  int i = blockIdx.x * blockDim.x + threadIdx.x;
  if (i >= NN * 64) return;
  int c = i & 63;
  float x = (h[i] - m[c]) * rsqrtf(v[c] + 1e-5f) * g[c] + b[c];
  x = fmaxf(x, 0.f);
  h[i] = x; h16[i] = (_Float16)x;
}

// ---------------------------------------------------------------------------
// GemNet edge featurizer: ea[E,52] (f16) = [edge_attr(4), rbf(16), trip(32)]
// trip = silu(g @ W1 + b1) @ W2 + b2, g = [rbf(16), ang(8)] padded to K=32
// ---------------------------------------------------------------------------
__global__ void k_gem(const float* __restrict__ pos, const float* __restrict__ eattr,
                      const float* __restrict__ ndir, const int* __restrict__ src,
                      const int* __restrict__ dst, const float* __restrict__ w1,
                      const float* __restrict__ b1, const float* __restrict__ w2,
                      const float* __restrict__ b2, _Float16* __restrict__ ea, int E_) {
  __shared__ _Float16 sW1[32 * 64];
  __shared__ _Float16 sW2[64 * 32];
  __shared__ _Float16 sA[4 * 16 * 32];
  __shared__ _Float16 sH[4 * 16 * 64];
  __shared__ float sd[64], sth[64];
  int tid = threadIdx.x, wave = tid >> 5, lane = tid & 31;
  for (int i = tid; i < 32 * 64; i += 128) {
    int k = i >> 6, c = i & 63;
    sW1[i] = (k < 24) ? (_Float16)w1[k * 64 + c] : (_Float16)0.f;
  }
  for (int i = tid; i < 64 * 32; i += 128) {
    int k = i >> 5, c = i & 31;
    sW2[i] = (_Float16)w2[k * 32 + c];
  }
  int ebase = (blockIdx.x * 4 + wave) * 16;
  if (lane < 16) {
    int e = ebase + lane; float dd = 1.f, th = 0.f;
    if (e < E_) {
      int s = src[e], dn = dst[e];
      float vx = pos[dn * 3 + 0] - pos[s * 3 + 0];
      float vy = pos[dn * 3 + 1] - pos[s * 3 + 1];
      float vz = pos[dn * 3 + 2] - pos[s * 3 + 2];
      dd = sqrtf(vx * vx + vy * vy + vz * vz + 1e-12f);
      float inv = 1.f / dd;
      float ca = (vx * ndir[s * 3] + vy * ndir[s * 3 + 1] + vz * ndir[s * 3 + 2]) * inv;
      ca = fminf(0.999f, fmaxf(-0.999f, ca));
      th = acosf(ca);
    }
    sd[wave * 16 + lane] = dd; sth[wave * 16 + lane] = th;
  }
  __syncthreads();
  _Float16* myA = sA + wave * (16 * 32);
  const float cstep = 4.0f / 15.0f;      // linspace(0,RMAX,16) spacing
  for (int i = lane; i < 16 * 32; i += 32) {
    int row = i >> 5, k = i & 31;
    float v = 0.f;
    if (k < 16) { float t = (sd[wave * 16 + row] - cstep * k) * 4.0f; v = expf(-t * t); }
    else if (k < 24) v = cosf(sth[wave * 16 + row] * (float)(k - 16));
    myA[i] = (_Float16)v;
  }
  __syncthreads();
  int m = lane & 15, hi = lane >> 4;
  _Float16* myH = sH + wave * (16 * 64);
#pragma unroll
  for (int t = 0; t < 4; ++t) {
    v8f c = {0.f, 0.f, 0.f, 0.f, 0.f, 0.f, 0.f, 0.f};
    v16h a = afrag_lds(myA, 32, m, hi, 0);
    v16h b = bfrag_lds(sW1, 64, t * 16, lane, 0);
    c = WMMA(a, b, c);
    int n = t * 16 + m;
    float bb = b1[n];
#pragma unroll
    for (int r = 0; r < 8; ++r) {
      float xv = c[r] + bb;
      myH[(size_t)(r + 8 * hi) * 64 + n] = (_Float16)(xv * sigf(xv));  // silu
    }
  }
  __syncthreads();
#pragma unroll
  for (int t = 0; t < 2; ++t) {
    v8f c = {0.f, 0.f, 0.f, 0.f, 0.f, 0.f, 0.f, 0.f};
#pragma unroll
    for (int ks = 0; ks < 2; ++ks) {
      v16h a = afrag_lds(myH, 64, m, hi, ks * 32);
      v16h b = bfrag_lds(sW2, 32, t * 16, lane, ks * 32);
      c = WMMA(a, b, c);
    }
    int n = t * 16 + m;
    float bb = b2[n];
#pragma unroll
    for (int r = 0; r < 8; ++r) {
      int e = ebase + r + 8 * hi;
      if (e < E_) ea[(size_t)e * 52 + 20 + n] = (_Float16)(c[r] + bb);
    }
  }
  for (int i = lane; i < 16 * 20; i += 32) {
    int row = i / 20, k = i - row * 20;
    int e = ebase + row;
    if (e < E_) {
      float v;
      if (k < 4) v = eattr[(size_t)e * 4 + k];
      else { float t = (sd[wave * 16 + row] - cstep * (k - 4)) * 4.0f; v = expf(-t * t); }
      ea[(size_t)e * 52 + k] = (_Float16)v;
    }
  }
}

// ---------------------------------------------------------------------------
// Edge attention logits: relu([x_src,x_dst,ea] @ W1 + b1) . w2 + b2, / TEMP
// ---------------------------------------------------------------------------
template <int CIN, int KPAD>
__global__ void k_att(const float* __restrict__ xn, const _Float16* __restrict__ ea,
                      const int* __restrict__ src, const int* __restrict__ dst,
                      const float* __restrict__ w1, const float* __restrict__ b1,
                      const float* __restrict__ w2, const float* __restrict__ b2,
                      float* __restrict__ logit, int E_) {
  constexpr int IN = 2 * CIN + 52;
  static_assert(KPAD >= IN && KPAD % 32 == 0, "bad KPAD");
  __shared__ _Float16 sB[KPAD * 64];
  __shared__ _Float16 sA[4 * 16 * KPAD];
  __shared__ int sid[4 * 32];
  int tid = threadIdx.x, wave = tid >> 5, lane = tid & 31;
  for (int i = tid; i < KPAD * 64; i += 128) {
    int k = i >> 6, c = i & 63;
    sB[i] = (k < IN) ? (_Float16)w1[(size_t)k * 64 + c] : (_Float16)0.f;
  }
  int ebase = (blockIdx.x * 4 + wave) * 16;
  if (lane < 16) {
    int e = ebase + lane;
    sid[wave * 32 + lane] = (e < E_) ? src[e] : 0;
    sid[wave * 32 + 16 + lane] = (e < E_) ? dst[e] : 0;
  }
  __syncthreads();
  _Float16* myA = sA + (size_t)wave * 16 * KPAD;
  const int* mid = sid + wave * 32;
  for (int i = lane; i < 16 * KPAD; i += 32) {
    int row = i / KPAD, k = i - row * KPAD;
    int e = ebase + row;
    float v = 0.f;
    if (e < E_) {
      if (k < CIN) v = xn[(size_t)mid[row] * CIN + k];
      else if (k < 2 * CIN) v = xn[(size_t)mid[16 + row] * CIN + (k - CIN)];
      else if (k < IN) v = (float)ea[(size_t)e * 52 + (k - 2 * CIN)];
    }
    myA[i] = (_Float16)v;
  }
  __syncthreads();
  int m = lane & 15, hi = lane >> 4;
  float w2r[4], b1r[4];
#pragma unroll
  for (int t = 0; t < 4; ++t) { int n = t * 16 + m; w2r[t] = w2[n]; b1r[t] = b1[n]; }
  float part[8];
#pragma unroll
  for (int r = 0; r < 8; ++r) part[r] = 0.f;
#pragma unroll
  for (int t = 0; t < 4; ++t) {
    v8f c = {0.f, 0.f, 0.f, 0.f, 0.f, 0.f, 0.f, 0.f};
#pragma unroll
    for (int ks = 0; ks < KPAD / 32; ++ks) {
      v16h a = afrag_lds(myA, KPAD, m, hi, ks * 32);
      v16h b = bfrag_lds(sB, 64, t * 16, lane, ks * 32);
      c = WMMA(a, b, c);
    }
#pragma unroll
    for (int r = 0; r < 8; ++r) {
      float hv = fmaxf(c[r] + b1r[t], 0.f);
      part[r] += hv * w2r[t];
    }
  }
  float b2v = b2[0];
#pragma unroll
  for (int r = 0; r < 8; ++r) {
    float v = part[r];
    v += __shfl_xor(v, 8, 16);
    v += __shfl_xor(v, 4, 16);
    v += __shfl_xor(v, 2, 16);
    v += __shfl_xor(v, 1, 16);
    if ((lane & 15) == 0) {
      int e = ebase + r + 8 * hi;
      if (e < E_) logit[e] = (v + b2v) * 1.25f;  // / TEMP(0.8)
    }
  }
}

// ---------------------------------------------------------------------------
// Gate: gate[E,64] f16 = sigmoid(ea @ We + be) * aE[e]
// ---------------------------------------------------------------------------
__global__ void k_gate(const _Float16* __restrict__ ea, const float* __restrict__ we,
                       const float* __restrict__ be, const float* __restrict__ aE,
                       _Float16* __restrict__ gate, int E_) {
  __shared__ _Float16 sW[64 * 64];
  __shared__ _Float16 sA[4 * 16 * 64];
  int tid = threadIdx.x, wave = tid >> 5, lane = tid & 31;
  for (int i = tid; i < 64 * 64; i += 128) {
    int k = i >> 6, c = i & 63;
    sW[i] = (k < 52) ? (_Float16)we[(size_t)k * 64 + c] : (_Float16)0.f;
  }
  int ebase = (blockIdx.x * 4 + wave) * 16;
  _Float16* myA = sA + (size_t)wave * 16 * 64;
  for (int i = lane; i < 16 * 64; i += 32) {
    int row = i >> 6, k = i & 63;
    int e = ebase + row;
    _Float16 v = (_Float16)0.f;
    if (e < E_ && k < 52) v = ea[(size_t)e * 52 + k];
    myA[i] = v;
  }
  __syncthreads();
  int m = lane & 15, hi = lane >> 4;
#pragma unroll
  for (int t = 0; t < 4; ++t) {
    v8f c = {0.f, 0.f, 0.f, 0.f, 0.f, 0.f, 0.f, 0.f};
#pragma unroll
    for (int ks = 0; ks < 2; ++ks) {
      v16h a = afrag_lds(myA, 64, m, hi, ks * 32);
      v16h b = bfrag_lds(sW, 64, t * 16, lane, ks * 32);
      c = WMMA(a, b, c);
    }
    int n = t * 16 + m;
    float bb = be[n];
#pragma unroll
    for (int r = 0; r < 8; ++r) {
      int e = ebase + r + 8 * hi;
      if (e < E_) gate[(size_t)e * 64 + n] = (_Float16)(sigf(c[r] + bb) * aE[e]);
    }
  }
}

// ---------------------------------------------------------------------------
// hW = h16 @ W  (N x 64 x 64)
// ---------------------------------------------------------------------------
__global__ void k_nodemm(const _Float16* __restrict__ h16, const float* __restrict__ W,
                         float* __restrict__ hW, int NN) {
  __shared__ _Float16 sW[64 * 64];
  int tid = threadIdx.x, wave = tid >> 5, lane = tid & 31;
  for (int i = tid; i < 64 * 64; i += 128) sW[i] = (_Float16)W[i];
  __syncthreads();
  int nodebase = (blockIdx.x * 4 + wave) * 16;
  int m = lane & 15, hi = lane >> 4;
  int arow = nodebase + m; if (arow >= NN) arow = NN - 1;
  const _Float16* rowp = h16 + (size_t)arow * 64;
  v16h a0 = afrag_row16(rowp, hi, 0);
  v16h a1 = afrag_row16(rowp, hi, 32);
#pragma unroll
  for (int t = 0; t < 4; ++t) {
    v8f c = {0.f, 0.f, 0.f, 0.f, 0.f, 0.f, 0.f, 0.f};
    c = WMMA(a0, bfrag_lds(sW, 64, t * 16, lane, 0), c);
    c = WMMA(a1, bfrag_lds(sW, 64, t * 16, lane, 32), c);
    int n = t * 16 + m;
#pragma unroll
    for (int r = 0; r < 8; ++r) {
      int node = nodebase + r + 8 * hi;
      if (node < NN) hW[(size_t)node * 64 + n] = c[r];
    }
  }
}

// ---------------------------------------------------------------------------
// GRU: gi = agg @ wih, gh = h @ whh  (N x 64 x 192 twice), gate fusion
// ---------------------------------------------------------------------------
__global__ void k_gru(const float* __restrict__ agg, float* __restrict__ h,
                      _Float16* __restrict__ h16, const float* __restrict__ wih,
                      const float* __restrict__ whh, const float* __restrict__ bih,
                      const float* __restrict__ bhh, int NN) {
  __shared__ _Float16 sWi[64 * 192];
  __shared__ _Float16 sWh[64 * 192];
  int tid = threadIdx.x, wave = tid >> 5, lane = tid & 31;
  for (int i = tid; i < 64 * 192; i += 128) {
    sWi[i] = (_Float16)wih[i];
    sWh[i] = (_Float16)whh[i];
  }
  __syncthreads();
  int nodebase = (blockIdx.x * 4 + wave) * 16;
  int m = lane & 15, hi = lane >> 4;
  int arow = nodebase + m; if (arow >= NN) arow = NN - 1;
  const float* ar = agg + (size_t)arow * 64;
  const float* hr = h + (size_t)arow * 64;
  v16h aA[2], hA[2];
#pragma unroll
  for (int ks = 0; ks < 2; ++ks) {
    aA[ks] = afrag_row32(ar, hi, ks * 32);
    hA[ks] = afrag_row32(hr, hi, ks * 32);
  }
#pragma unroll
  for (int t = 0; t < 4; ++t) {
    v8f cR = {0.f,0.f,0.f,0.f,0.f,0.f,0.f,0.f}, cZ = cR, cN = cR;
    v8f dR = cR, dZ = cR, dN = cR;
#pragma unroll
    for (int ks = 0; ks < 2; ++ks) {
      cR = WMMA(aA[ks], bfrag_lds(sWi, 192, (t    ) * 16, lane, ks * 32), cR);
      cZ = WMMA(aA[ks], bfrag_lds(sWi, 192, (t + 4) * 16, lane, ks * 32), cZ);
      cN = WMMA(aA[ks], bfrag_lds(sWi, 192, (t + 8) * 16, lane, ks * 32), cN);
      dR = WMMA(hA[ks], bfrag_lds(sWh, 192, (t    ) * 16, lane, ks * 32), dR);
      dZ = WMMA(hA[ks], bfrag_lds(sWh, 192, (t + 4) * 16, lane, ks * 32), dZ);
      dN = WMMA(hA[ks], bfrag_lds(sWh, 192, (t + 8) * 16, lane, ks * 32), dN);
    }
    int n0 = t * 16 + m;
    float biR = bih[n0], biZ = bih[64 + n0], biN = bih[128 + n0];
    float bhR = bhh[n0], bhZ = bhh[64 + n0], bhN = bhh[128 + n0];
#pragma unroll
    for (int r = 0; r < 8; ++r) {
      int node = nodebase + r + 8 * hi;
      int nc = (node < NN) ? node : NN - 1;
      float hold = h[(size_t)nc * 64 + n0];
      float rr = sigf(cR[r] + biR + dR[r] + bhR);
      float zz = sigf(cZ[r] + biZ + dZ[r] + bhZ);
      float nn = tanhf(cN[r] + biN + rr * (dN[r] + bhN));
      float hv = (1.f - zz) * nn + zz * hold;
      if (node < NN) {
        h[(size_t)node * 64 + n0] = hv;
        h16[(size_t)node * 64 + n0] = (_Float16)hv;
      }
    }
  }
}

// ---------------------------------------------------------------------------
// Set2Set
// ---------------------------------------------------------------------------
__global__ void k_lstm(const float* wih0, const float* whh0, const float* bih0, const float* bhh0,
                       const float* wih1, const float* whh1, const float* bih1, const float* bhh1,
                       const float* __restrict__ qs, float* h0, float* c0, float* h1, float* c1,
                       int B_) {
  int b = threadIdx.x;
  if (b >= B_) return;
  float h0n[64];
  for (int j = 0; j < 64; ++j) {
    float gi = bih0[j] + bhh0[j];
    float gf = bih0[64 + j] + bhh0[64 + j];
    float gg = bih0[128 + j] + bhh0[128 + j];
    float go = bih0[192 + j] + bhh0[192 + j];
    for (int k = 0; k < 128; ++k) {
      float xv = qs[(size_t)b * 128 + k];
      gi += xv * wih0[(size_t)k * 256 + j];
      gf += xv * wih0[(size_t)k * 256 + 64 + j];
      gg += xv * wih0[(size_t)k * 256 + 128 + j];
      go += xv * wih0[(size_t)k * 256 + 192 + j];
    }
    for (int k = 0; k < 64; ++k) {
      float hv = h0[(size_t)b * 64 + k];
      gi += hv * whh0[(size_t)k * 256 + j];
      gf += hv * whh0[(size_t)k * 256 + 64 + j];
      gg += hv * whh0[(size_t)k * 256 + 128 + j];
      go += hv * whh0[(size_t)k * 256 + 192 + j];
    }
    float cc = sigf(gf) * c0[(size_t)b * 64 + j] + sigf(gi) * tanhf(gg);
    c0[(size_t)b * 64 + j] = cc;
    h0n[j] = sigf(go) * tanhf(cc);
  }
  float h1n[64];
  for (int j = 0; j < 64; ++j) {
    float gi = bih1[j] + bhh1[j];
    float gf = bih1[64 + j] + bhh1[64 + j];
    float gg = bih1[128 + j] + bhh1[128 + j];
    float go = bih1[192 + j] + bhh1[192 + j];
    for (int k = 0; k < 64; ++k) {
      float xv = h0n[k];
      gi += xv * wih1[(size_t)k * 256 + j];
      gf += xv * wih1[(size_t)k * 256 + 64 + j];
      gg += xv * wih1[(size_t)k * 256 + 128 + j];
      go += xv * wih1[(size_t)k * 256 + 192 + j];
    }
    for (int k = 0; k < 64; ++k) {
      float hv = h1[(size_t)b * 64 + k];
      gi += hv * whh1[(size_t)k * 256 + j];
      gf += hv * whh1[(size_t)k * 256 + 64 + j];
      gg += hv * whh1[(size_t)k * 256 + 128 + j];
      go += hv * whh1[(size_t)k * 256 + 192 + j];
    }
    float cc = sigf(gf) * c1[(size_t)b * 64 + j] + sigf(gi) * tanhf(gg);
    c1[(size_t)b * 64 + j] = cc;
    h1n[j] = sigf(go) * tanhf(cc);
  }
  for (int j = 0; j < 64; ++j) {
    h0[(size_t)b * 64 + j] = h0n[j];
    h1[(size_t)b * 64 + j] = h1n[j];
  }
}

__global__ void k_s2s_reset(unsigned* mordB, float* sB, float* racc, int B_) {
  int i = blockIdx.x * blockDim.x + threadIdx.x;
  if (i >= B_ * 64) return;
  racc[i] = 0.f;
  if (i < B_) { mordB[i] = ford(-3.0e38f); sB[i] = 0.f; }
}

__global__ void k_energy(const float* __restrict__ h, const float* __restrict__ h1,
                         const int* __restrict__ batch, float* __restrict__ eN,
                         unsigned* mordB, int NN) {
  int n = blockIdx.x * blockDim.x + threadIdx.x;
  if (n >= NN) return;
  int b = batch[n];
  float e = 0.f;
  for (int c = 0; c < 64; ++c) e += h[(size_t)n * 64 + c] * h1[(size_t)b * 64 + c];
  eN[n] = e;
  atomicMax(&mordB[b], ford(e));
}

__global__ void k_s2s_accum(const float* __restrict__ h, const float* __restrict__ eN,
                            const int* __restrict__ batch, const unsigned* __restrict__ mordB,
                            float* sB, float* racc, int NN) {
  int idx = blockIdx.x * blockDim.x + threadIdx.x;
  if (idx >= NN * 64) return;
  int n = idx >> 6, c = idx & 63;
  int b = batch[n];
  float w = expf(eN[n] - fordinv(mordB[b]));
  if (c == 0) atomicAdd(&sB[b], w);
  atomicAdd(&racc[(size_t)b * 64 + c], w * h[idx]);
}

__global__ void k_qs(const float* __restrict__ h1, const float* __restrict__ racc,
                     const float* __restrict__ sB, float* __restrict__ qs, int B_) {
  int i = blockIdx.x * blockDim.x + threadIdx.x;
  if (i >= B_ * 64) return;
  int b = i >> 6, c = i & 63;
  qs[(size_t)b * 128 + c] = h1[i];
  qs[(size_t)b * 128 + 64 + c] = racc[i] / (sB[b] + 1e-16f);
}

__global__ void k_meanacc(const float* __restrict__ h, const int* __restrict__ batch,
                          float* msum, float* cnt, int NN) {
  int idx = blockIdx.x * blockDim.x + threadIdx.x;
  if (idx >= NN * 64) return;
  int n = idx >> 6, c = idx & 63;
  int b = batch[n];
  atomicAdd(&msum[(size_t)b * 64 + c], h[idx]);
  if (c == 0) atomicAdd(&cnt[b], 1.f);
}

__global__ void k_head(const float* __restrict__ qs, const float* __restrict__ msum,
                       const float* __restrict__ cnt,
                       const float* pbb, const float* pbg, const float* pbm, const float* pbv,
                       const float* fw0, const float* fb0,
                       const float* g0, const float* bb0, const float* m0, const float* v0,
                       const float* fw1, const float* fb1,
                       const float* g1, const float* bb1, const float* m1, const float* v1,
                       const float* fw2, const float* fb2, float* out, int B_) {
  int b = blockIdx.x * blockDim.x + threadIdx.x;
  if (b >= B_) return;
  float z[192];
  float invc = 1.f / fmaxf(cnt[b], 1.f);
  for (int i = 0; i < 128; ++i) z[i] = qs[(size_t)b * 128 + i];
  for (int i = 0; i < 64; ++i) z[128 + i] = msum[(size_t)b * 64 + i] * invc;
  for (int i = 0; i < 192; ++i) z[i] = (z[i] - pbm[i]) * rsqrtf(pbv[i] + 1e-5f) * pbg[i] + pbb[i];
  float z1[96];
  for (int j = 0; j < 96; ++j) {
    float s = fb0[j];
    for (int k = 0; k < 192; ++k) s += z[k] * fw0[(size_t)k * 96 + j];
    s = (s - m0[j]) * rsqrtf(v0[j] + 1e-5f) * g0[j] + bb0[j];
    z1[j] = fmaxf(s, 0.f);
  }
  float z2[48];
  for (int j = 0; j < 48; ++j) {
    float s = fb1[j];
    for (int k = 0; k < 96; ++k) s += z1[k] * fw1[(size_t)k * 48 + j];
    s = (s - m1[j]) * rsqrtf(v1[j] + 1e-5f) * g1[j] + bb1[j];
    z2[j] = fmaxf(s, 0.f);
  }
  float s = fb2[0];
  for (int k = 0; k < 48; ++k) s += z2[k] * fw2[k];
  out[b] = s;
}

// ---------------------------------------------------------------------------
// host launcher
// ---------------------------------------------------------------------------
extern "C" void kernel_launch(void* const* d_in, const int* in_sizes, int n_in,
                              void* d_out, int out_size, void* d_ws, size_t ws_size,
                              hipStream_t stream) {
  const int N = in_sizes[0] / 4;
  const int E = in_sizes[2] / 4;
  const int B = out_size;
  const float* x = (const float*)d_in[0];
  const float* pos = (const float*)d_in[1];
  const float* eattr = (const float*)d_in[2];
  // params leaves in jax tree-flatten order (dict keys sorted at every level)
  int p = 3;
  const float *a_l1b[3], *a_l1w[3], *a_l2b[3], *a_l2w[3];
  for (int i = 0; i < 3; ++i) {
    a_l1b[i] = (const float*)d_in[p++]; a_l1w[i] = (const float*)d_in[p++];
    a_l2b[i] = (const float*)d_in[p++]; a_l2w[i] = (const float*)d_in[p++];
  }
  const float *bn_b[3], *bn_g[3], *bn_m[3], *bn_v[3];
  for (int i = 0; i < 3; ++i) {
    bn_b[i] = (const float*)d_in[p++]; bn_g[i] = (const float*)d_in[p++];
    bn_m[i] = (const float*)d_in[p++]; bn_v[i] = (const float*)d_in[p++];
  }
  const float *cW[3], *ceb[3], *cew[3], *cbhh[3], *cbih[3], *cwhh[3], *cwih[3];
  for (int i = 0; i < 3; ++i) {
    cW[i]   = (const float*)d_in[p++]; ceb[i]  = (const float*)d_in[p++];
    cew[i]  = (const float*)d_in[p++]; cbhh[i] = (const float*)d_in[p++];
    cbih[i] = (const float*)d_in[p++]; cwhh[i] = (const float*)d_in[p++];
    cwih[i] = (const float*)d_in[p++];
  }
  const float *fcb[3], *fcw[3];
  for (int i = 0; i < 3; ++i) { fcb[i] = (const float*)d_in[p++]; fcw[i] = (const float*)d_in[p++]; }
  const float *fbb[2], *fbg[2], *fbm[2], *fbv[2];
  for (int i = 0; i < 2; ++i) {
    fbb[i] = (const float*)d_in[p++]; fbg[i] = (const float*)d_in[p++];
    fbm[i] = (const float*)d_in[p++]; fbv[i] = (const float*)d_in[p++];
  }
  const float* g_l1b = (const float*)d_in[p++]; const float* g_l1w = (const float*)d_in[p++];
  const float* g_l2b = (const float*)d_in[p++]; const float* g_l2w = (const float*)d_in[p++];
  const float* l0bhh = (const float*)d_in[p++]; const float* l0bih = (const float*)d_in[p++];
  const float* l0whh = (const float*)d_in[p++]; const float* l0wih = (const float*)d_in[p++];
  const float* l1bhh = (const float*)d_in[p++]; const float* l1bih = (const float*)d_in[p++];
  const float* l1whh = (const float*)d_in[p++]; const float* l1wih = (const float*)d_in[p++];
  const float* pbb = (const float*)d_in[p++]; const float* pbg = (const float*)d_in[p++];
  const float* pbm = (const float*)d_in[p++]; const float* pbv = (const float*)d_in[p++];
  const int* ei = (const int*)d_in[p++];
  const int* batch = (const int*)d_in[p++];
  const int* src = ei;
  const int* dst = ei + E;
  (void)n_in; (void)ws_size;

  // workspace carve (~240 MB total)
  uintptr_t cur = (uintptr_t)d_ws;
  auto take = [&](size_t bytes) -> void* {
    void* r = (void*)cur;
    cur += (bytes + 255) & ~(size_t)255;
    return r;
  };
  float*    ndir  = (float*)take((size_t)N * 3 * 4);
  _Float16* ea    = (_Float16*)take((size_t)E * 52 * 2);
  float*    logit = (float*)take((size_t)E * 4);
  float*    aE    = (float*)take((size_t)E * 4);
  unsigned* mord  = (unsigned*)take((size_t)N * 4);
  float*    ssum  = (float*)take((size_t)N * 4);
  float*    h     = (float*)take((size_t)N * 64 * 4);
  _Float16* h16   = (_Float16*)take((size_t)N * 64 * 2);
  float*    hW    = (float*)take((size_t)N * 64 * 4);
  float*    agg   = (float*)take((size_t)N * 64 * 4);
  _Float16* gate  = (_Float16*)take((size_t)E * 64 * 2);
  float*    eN    = (float*)take((size_t)N * 4);
  float*    qs    = (float*)take((size_t)B * 128 * 4);
  float*    h0    = (float*)take((size_t)B * 64 * 4);
  float*    c0    = (float*)take((size_t)B * 64 * 4);
  float*    h1    = (float*)take((size_t)B * 64 * 4);
  float*    c1    = (float*)take((size_t)B * 64 * 4);
  unsigned* mordB = (unsigned*)take((size_t)B * 4);
  float*    sB    = (float*)take((size_t)B * 4);
  float*    racc  = (float*)take((size_t)B * 64 * 4);
  float*    cnt   = (float*)take((size_t)B * 4);
  float*    msum  = (float*)take((size_t)B * 64 * 4);

  const int T = 256;
  auto cdiv = [](int a, int b) { return (a + b - 1) / b; };
  int gE = cdiv(E, T), gN = cdiv(N, T);
  int gE64 = cdiv(E * 64, T), gN64 = cdiv(N * 64, T);
  int gEw = cdiv(E, 64), gNw = cdiv(N, 64);

  // geometry + GemNet edge features
  k_zero<<<cdiv(N * 3, T), T, 0, stream>>>(ndir, N * 3);
  k_ndir_accum<<<gE, T, 0, stream>>>(pos, src, dst, ndir, E);
  k_ndir_norm<<<gN, T, 0, stream>>>(ndir, N);
  k_gem<<<gEw, 128, 0, stream>>>(pos, eattr, ndir, src, dst, g_l1w, g_l1b, g_l2w, g_l2b, ea, E);
  k_init_h<<<gN64, T, 0, stream>>>(x, h, h16, N);

  for (int l = 0; l < 3; ++l) {
    if (l == 0)
      k_att<4, 64><<<gEw, 128, 0, stream>>>(x, ea, src, dst, a_l1w[0], a_l1b[0], a_l2w[0],
                                            a_l2b[0], logit, E);
    else
      k_att<64, 192><<<gEw, 128, 0, stream>>>(h, ea, src, dst, a_l1w[l], a_l1b[l], a_l2w[l],
                                              a_l2b[l], logit, E);
    k_smax_init<<<gN, T, 0, stream>>>(mord, ssum, N);
    k_smax1<<<gE, T, 0, stream>>>(logit, dst, mord, E);
    k_smax2<<<gE, T, 0, stream>>>(logit, dst, mord, ssum, aE, E);
    k_smax3<<<gE, T, 0, stream>>>(dst, ssum, aE, E);
    k_gate<<<gEw, 128, 0, stream>>>(ea, cew[l], ceb[l], aE, gate, E);
    for (int s = 0; s < 3; ++s) {
      k_nodemm<<<gNw, 128, 0, stream>>>(h16, cW[l] + (size_t)s * 64 * 64, hW, N);
      k_zero<<<gN64, T, 0, stream>>>(agg, N * 64);
      k_message<<<gE64, T, 0, stream>>>(hW, gate, src, dst, agg, E);
      k_gru<<<gNw, 128, 0, stream>>>(agg, h, h16, cwih[l], cwhh[l], cbih[l], cbhh[l], N);
    }
    k_bnrelu<<<gN64, T, 0, stream>>>(h, h16, bn_g[l], bn_b[l], bn_m[l], bn_v[l], N);
  }

  // Set2Set pooling
  k_zero<<<cdiv(B * 128, T), T, 0, stream>>>(qs, B * 128);
  k_zero<<<cdiv(B * 64, T), T, 0, stream>>>(h0, B * 64);
  k_zero<<<cdiv(B * 64, T), T, 0, stream>>>(c0, B * 64);
  k_zero<<<cdiv(B * 64, T), T, 0, stream>>>(h1, B * 64);
  k_zero<<<cdiv(B * 64, T), T, 0, stream>>>(c1, B * 64);
  for (int it = 0; it < 5; ++it) {
    k_lstm<<<1, B, 0, stream>>>(l0wih, l0whh, l0bih, l0bhh, l1wih, l1whh, l1bih, l1bhh,
                                qs, h0, c0, h1, c1, B);
    k_s2s_reset<<<cdiv(B * 64, T), T, 0, stream>>>(mordB, sB, racc, B);
    k_energy<<<gN, T, 0, stream>>>(h, h1, batch, eN, mordB, N);
    k_s2s_accum<<<gN64, T, 0, stream>>>(h, eN, batch, mordB, sB, racc, N);
    k_qs<<<cdiv(B * 64, T), T, 0, stream>>>(h1, racc, sB, qs, B);
  }
  k_zero<<<cdiv(B * 64, T), T, 0, stream>>>(msum, B * 64);
  k_zero<<<cdiv(B, T), T, 0, stream>>>(cnt, B);
  k_meanacc<<<gN64, T, 0, stream>>>(h, batch, msum, cnt, N);
  k_head<<<cdiv(B, 256), 256, 0, stream>>>(qs, msum, cnt, pbb, pbg, pbm, pbv,
                                           fcw[0], fcb[0], fbg[0], fbb[0], fbm[0], fbv[0],
                                           fcw[1], fcb[1], fbg[1], fbb[1], fbm[1], fbv[1],
                                           fcw[2], fcb[2], (float*)d_out, B);
}
